// ProgramVectorizer_66030827209239
// MI455X (gfx1250) — matrix operand order repaired and checked by reading, not compiled, not run on hardware
//
#include <hip/hip_runtime.h>

typedef __attribute__((ext_vector_type(16))) _Float16 v16h;
typedef __attribute__((ext_vector_type(8)))  float    v8f;

#define DMODEL 128
#define W2_STRIDE 136   // halfs; padded so row accesses spread LDS banks
#define LN_EPS 1e-5f

// Branch-free exact-ish GELU: erf via Abramowitz-Stegun 7.1.26 (|err|~2.5e-5).
// No EXEC-divergent slow path, ~12 VALU ops, 1 v_rcp + 1 v_exp.
__device__ __forceinline__ float gelu_f(float x) {
    const float z = fabsf(x) * 0.70710678118654752f;
    const float t = __builtin_amdgcn_rcpf(__builtin_fmaf(0.3275911f, z, 1.0f));
    float p = __builtin_fmaf(t, 1.061405429f, -1.453152027f);
    p = __builtin_fmaf(t, p, 1.421413741f);
    p = __builtin_fmaf(t, p, -0.284496736f);
    p = __builtin_fmaf(t, p, 0.254829592f);
    const float erfa = __builtin_fmaf(-(p * t), __expf(-z * z), 1.0f);
    return 0.5f * x * (1.0f + copysignf(erfa, x));
}

__global__ __launch_bounds__(256)
void progvec_wmma_kernel(
    const int* __restrict__ type_id,        const int* __restrict__ symbol_idx,
    const int* __restrict__ field_idx,      const int* __restrict__ family_idx,
    const int* __restrict__ entity_type_idx,const int* __restrict__ entity_id,
    const int* __restrict__ dim_idx,        const int* __restrict__ pos_idx,
    const float* __restrict__ values,
    const float* __restrict__ symbol_emb,   const float* __restrict__ field_emb,
    const float* __restrict__ family_emb,   const float* __restrict__ entity_type_emb,
    const float* __restrict__ entity_id_emb,const float* __restrict__ dim_emb,
    const float* __restrict__ index_pos_emb,
    const float* __restrict__ w1,  const float* __restrict__ b1,
    const float* __restrict__ w2,  const float* __restrict__ b2,
    const float* __restrict__ ln_gamma, const float* __restrict__ ln_beta,
    float* __restrict__ out, int n_tiles)
{
    __shared__ alignas(16) _Float16 w2h[DMODEL * W2_STRIDE];
    __shared__ alignas(16) float    w1s[DMODEL];
    __shared__ alignas(16) float    b1s[DMODEL];

    // ---- stage w2 (f32 -> f16, padded) and w1/b1 into LDS ----
    for (int i = threadIdx.x; i < DMODEL * DMODEL; i += blockDim.x) {
        int n = i >> 7, k = i & 127;
        w2h[n * W2_STRIDE + k] = (_Float16)w2[i];
    }
    for (int i = threadIdx.x; i < DMODEL; i += blockDim.x) {
        w1s[i] = w1[i];
        b1s[i] = b1[i];
    }
    __syncthreads();

    const int  lane   = threadIdx.x & 31;
    const int  wave   = threadIdx.x >> 5;
    const int  lanelo = lane & 15;
    const bool hiHalf = lane >= 16;
    const int  koffA  = hiHalf ? 8  : 0;   // A fragment K pattern (16-bit 16x32)
    const int  koffB  = hiHalf ? 8  : 0;   // B fragment K pattern (16-bit 32x16)

    // Per-lane tile-invariant vectors: d = c*16 + lanelo for column tile c
    float b2v[8], gv[8], bvv[8];
#pragma unroll
    for (int c = 0; c < 8; ++c) {
        const int d = c * 16 + lanelo;
        b2v[c] = b2[d];
        gv[c]  = ln_gamma[d];
        bvv[c] = ln_beta[d];
    }

    const long waveStride = (long)gridDim.x * 8;
    for (long tile = (long)blockIdx.x * 8 + wave; tile < n_tiles; tile += waveStride) {
        const long t0 = tile * 16;

        // Opaque ZERO index, refreshed each iteration: defeats LICM of the
        // tile-invariant LDS loads (which otherwise hoist + spill to scratch),
        // while keeping the pointer chain rooted at the __shared__ arrays so
        // address-space inference still selects ds_load_b128 (not flat_load).
        int zofs = 0;
        asm volatile("" : "+v"(zofs));
        const _Float16* w2p = w2h + zofs;
        const float*    w1p = w1s + zofs;
        const float*    b1p = b1s + zofs;

        // ---- value transform for this lane's A-row (token t0 + lanelo) ----
        const float vraw = values[t0 + lanelo];
        const float vlog = copysignf(log1pf(fabsf(vraw)), vraw);

        // ---- GEMM, K-step outer so only one A fragment is live at a time ----
        v8f acc[8] = {};
#pragma unroll
        for (int ks = 0; ks < 4; ++ks) {
            const int kb = ks * 32 + koffA;

            // lane's 16 w1/b1 values: two contiguous 8-float runs -> 4 float4 each
            union F16x { float4 v4[4]; float f[16]; } W, Bz;
            W.v4[0]  = ((const float4*)(w1p + kb))[0];
            W.v4[1]  = ((const float4*)(w1p + kb))[1];
            W.v4[2]  = ((const float4*)(w1p + kb + 16))[0];
            W.v4[3]  = ((const float4*)(w1p + kb + 16))[1];
            Bz.v4[0] = ((const float4*)(b1p + kb))[0];
            Bz.v4[1] = ((const float4*)(b1p + kb))[1];
            Bz.v4[2] = ((const float4*)(b1p + kb + 16))[0];
            Bz.v4[3] = ((const float4*)(b1p + kb + 16))[1];

            // A fragment element e maps 1:1 onto packed element e
            v16h Afrag;
#pragma unroll
            for (int e = 0; e < 16; ++e) {
                const float x = __builtin_fmaf(vlog, W.f[e], Bz.f[e]);
                Afrag[e] = (_Float16)gelu_f(x);
            }

#pragma unroll
            for (int c = 0; c < 8; ++c) {
                const int n = c * 16 + lanelo;                 // output dim of this lane
                const _Float16* bp = w2p + n * W2_STRIDE + koffB + ks * 32;
                union { v16h v; uint4 q[2]; } Bf;
                Bf.q[0] = ((const uint4*)bp)[0];
                Bf.q[1] = ((const uint4*)bp)[2];
                acc[c] = __builtin_amdgcn_wmma_f32_16x16x32_f16(
                    false, Afrag, false, Bf.v, (short)0, acc[c], false, false);
            }
        }

        // ---- embeddings + masked MLP + LayerNorm + store, per C row ----
#pragma unroll
        for (int r = 0; r < 8; ++r) {
            const long t   = t0 + r + (hiHalf ? 8 : 0);
            const int  tid = type_id[t];
            const long sI   = symbol_idx[t];
            const long fI   = field_idx[t];
            const long famI = family_idx[t];
            const long etI  = entity_type_idx[t];
            const long eI   = entity_id[t];
            const long dI   = dim_idx[t];
            const long pI   = pos_idx[t];
            const bool msym = (tid == 0);
            const bool mdim = (tid == 2) | (tid == 3);
            const bool mpos = (tid == 3);
            const bool mval = (tid == 1) | (tid == 2);

#pragma unroll
            for (int c = 0; c < 8; ++c) {
                const int d = c * 16 + lanelo;
                float e;
                if (msym) {
                    e = symbol_emb[sI * DMODEL + d];
                } else {
                    e = field_emb[fI * DMODEL + d]
                      + family_emb[famI * DMODEL + d]
                      + entity_type_emb[etI * DMODEL + d]
                      + entity_id_emb[eI * DMODEL + d];
                }
                if (mdim) e += dim_emb[dI * DMODEL + d];
                if (mpos) e += index_pos_emb[pI * DMODEL + d];
                if (mval) e += acc[c][r] + b2v[c];
                acc[c][r] = e;
            }

            // LayerNorm over 128 dims: lane holds 8, reduce across the 16-lane half
            float s = 0.f;
#pragma unroll
            for (int c = 0; c < 8; ++c) s += acc[c][r];
            s += __shfl_xor(s, 1);
            s += __shfl_xor(s, 2);
            s += __shfl_xor(s, 4);
            s += __shfl_xor(s, 8);
            const float mu = s * (1.0f / 128.0f);

            float q = 0.f;
#pragma unroll
            for (int c = 0; c < 8; ++c) {
                const float dd = acc[c][r] - mu;
                q += dd * dd;
            }
            q += __shfl_xor(q, 1);
            q += __shfl_xor(q, 2);
            q += __shfl_xor(q, 4);
            q += __shfl_xor(q, 8);
            const float sc = rsqrtf(q * (1.0f / 128.0f) + LN_EPS);

            float* op = out + t * DMODEL;
#pragma unroll
            for (int c = 0; c < 8; ++c) {
                op[c * 16 + lanelo] = (acc[c][r] - mu) * sc * gv[c] + bvv[c];
            }
        }
    }
}

extern "C" void kernel_launch(void* const* d_in, const int* in_sizes, int n_in,
                              void* d_out, int out_size, void* d_ws, size_t ws_size,
                              hipStream_t stream) {
    const int*   type_id         = (const int*)d_in[0];
    const int*   symbol_idx      = (const int*)d_in[1];
    const int*   field_idx       = (const int*)d_in[2];
    const int*   family_idx      = (const int*)d_in[3];
    const int*   entity_type_idx = (const int*)d_in[4];
    const int*   entity_id       = (const int*)d_in[5];
    const int*   dim_idx         = (const int*)d_in[6];
    const int*   pos_idx         = (const int*)d_in[7];
    const float* values          = (const float*)d_in[8];
    const float* symbol_emb      = (const float*)d_in[9];
    const float* field_emb       = (const float*)d_in[10];
    const float* family_emb      = (const float*)d_in[11];
    const float* entity_type_emb = (const float*)d_in[12];
    const float* entity_id_emb   = (const float*)d_in[13];
    const float* dim_emb         = (const float*)d_in[14];
    const float* index_pos_emb   = (const float*)d_in[15];
    const float* w1              = (const float*)d_in[16];
    const float* b1              = (const float*)d_in[17];
    const float* w2              = (const float*)d_in[18];
    const float* b2              = (const float*)d_in[19];
    const float* ln_gamma        = (const float*)d_in[20];
    const float* ln_beta         = (const float*)d_in[21];
    float* out = (float*)d_out;

    const int N       = in_sizes[0];
    const int n_tiles = N / 16;       // N = 1,000,000 -> exactly 62500 tiles

    const int threads = 256;          // 8 waves/block, 1 tile per wave per step
    const int blocks  = 1024;
    progvec_wmma_kernel<<<blocks, threads, 0, stream>>>(
        type_id, symbol_idx, field_idx, family_idx, entity_type_idx, entity_id,
        dim_idx, pos_idx, values, symbol_emb, field_emb, family_emb,
        entity_type_emb, entity_id_emb, dim_emb, index_pos_emb,
        w1, b1, w2, b2, ln_gamma, ln_beta, out, n_tiles);
}
